// Embed_39170101740010
// MI455X (gfx1250) — matrix operand order repaired and checked by reading, not compiled, not run on hardware
//
#include <hip/hip_runtime.h>

#define LSEQ  16384
#define BATCH 64
#define KK    128      // number of kernels == padded tap count
#define NTILE 128      // output positions per workgroup

typedef __bf16  bf16_t;
typedef bf16_t  v16bf __attribute__((ext_vector_type(16)));
typedef bf16_t  v8bf  __attribute__((ext_vector_type(8)));
typedef float   v8f   __attribute__((ext_vector_type(8)));

union Frag { v16bf f; v8bf h[2]; };

struct WPtrs { const float* w[KK]; };

// ---- build zero-padded lower-triangular tap matrix, split into bf16 hi/lo ----
__global__ void build_w_ptrs(WPtrs wp, bf16_t* __restrict__ Wh, bf16_t* __restrict__ Wl) {
    int i = blockIdx.x;          // kernel index (size i+1)
    int t = threadIdx.x;         // tap
    float v = (t <= i) ? wp.w[i][t] : 0.0f;
    bf16_t hi = (bf16_t)v;
    bf16_t lo = (bf16_t)(v - (float)hi);
    Wh[i * KK + t] = hi;
    Wl[i * KK + t] = lo;
}

__global__ void build_w_concat(const float* __restrict__ wcat,
                               bf16_t* __restrict__ Wh, bf16_t* __restrict__ Wl) {
    int i = blockIdx.x;
    int t = threadIdx.x;
    int off = i * (i + 1) / 2;   // sum of sizes 1..i
    float v = (t <= i) ? wcat[off + t] : 0.0f;
    bf16_t hi = (bf16_t)v;
    bf16_t lo = (bf16_t)(v - (float)hi);
    Wh[i * KK + t] = hi;
    Wl[i * KK + t] = lo;
}

// ---- main GEMM-formulated conv kernel: C[128, NTILE] = W(128x128) * Hankel(x) ----
__global__ __launch_bounds__(256)
void conv_wmma(const float* __restrict__ x,
               const float* __restrict__ bias,
               const bf16_t* __restrict__ Wh,
               const bf16_t* __restrict__ Wl,
               float* __restrict__ out) {
    // 15 distinct B half-fragments (sliding 16-wide windows), stored in exact
    // WMMA-B per-lane layout: slot [w][n] = 16 consecutive bf16 (32B, 16B aligned)
    __shared__ __align__(16) bf16_t sBh[15 * 16 * 16];
    __shared__ __align__(16) bf16_t sBl[15 * 16 * 16];
    __shared__ float sx[NTILE + 128];        // raw fp32 x chunk (255 used)
    __shared__ __align__(16) float sbias[KK];
    __shared__ __align__(16) unsigned int sRow[KK];  // ragged row start + b*len (fits u32)

    const int tid = threadIdx.x;
    const int b   = blockIdx.y;
    const int j0  = blockIdx.x * NTILE;

    // stage x chunk (zero-pad past end of sequence)
    {
        int idx = j0 + tid;
        sx[tid] = (idx < LSEQ) ? x[(size_t)b * LSEQ + idx] : 0.0f;
    }
    if (tid < KK) {
        sbias[tid] = bias[tid];
        // off(i) = 64*(i*L - i(i-1)/2) + b*(L-i) = b*L + i*(64L - b) - 32*i*(i-1)
        unsigned int i = (unsigned int)tid;
        sRow[tid] = (unsigned int)b * LSEQ
                  + i * (unsigned int)(BATCH * LSEQ - b)
                  - 32u * i * (i - 1u);
    }
    __syncthreads();

    // expand the 15 windows into bf16 hi/lo B-fragment slots (240 slots)
    if (tid < 240) {
        int w = tid >> 4, n = tid & 15;
        bf16_t* dh = &sBh[(w * 16 + n) * 16];
        bf16_t* dl = &sBl[(w * 16 + n) * 16];
        #pragma unroll
        for (int e = 0; e < 16; ++e) {
            float v  = sx[w * 16 + n + e];
            bf16_t h = (bf16_t)v;
            dh[e] = h;
            dl[e] = (bf16_t)(v - (float)h);
        }
    }
    __syncthreads();

    const int lane  = tid & 31;
    const int wv    = tid >> 5;      // wave id -> M tile
    const int khalf = lane >> 4;     // fragment half (upper 16 lanes)
    const int nrow  = lane & 15;     // A: M row in tile / B,C,D: N column

    // hoist ALL A fragments (4 K-steps, hi+lo) -> one wait, pure LDS+WMMA loop
    Frag ah[4], al[4];
    {
        const bf16_t* wh = Wh + (wv * 16 + nrow) * KK + khalf * 16;
        const bf16_t* wl = Wl + (wv * 16 + nrow) * KK + khalf * 16;
        #pragma unroll
        for (int ks = 0; ks < 4; ++ks) {
            ah[ks].h[0] = *(const v8bf*)(wh + ks * 32);
            ah[ks].h[1] = *(const v8bf*)(wh + ks * 32 + 8);
            al[ks].h[0] = *(const v8bf*)(wl + ks * 32);
            al[ks].h[1] = *(const v8bf*)(wl + ks * 32 + 8);
        }
    }

    v8f acc[8];
    #pragma unroll
    for (int nt = 0; nt < 8; ++nt) acc[nt] = (v8f){};

    #pragma unroll
    for (int ks = 0; ks < 4; ++ks) {
        #pragma unroll
        for (int nt = 0; nt < 8; ++nt) {
            // Hankel structure: B fragment == window (nt + 2*ks + khalf)
            const int w = nt + 2 * ks + khalf;
            const v8bf* ph = (const v8bf*)&sBh[(w * 16 + nrow) * 16];
            const v8bf* pl = (const v8bf*)&sBl[(w * 16 + nrow) * 16];
            Frag bh, bl;
            bh.h[0] = ph[0]; bh.h[1] = ph[1];
            bl.h[0] = pl[0]; bl.h[1] = pl[1];
            // 3-term bf16 split ~= fp32 GEMM
            acc[nt] = __builtin_amdgcn_wmma_f32_16x16x32_bf16(
                          false, ah[ks].f, false, bh.f, (short)0, acc[nt], false, false);
            acc[nt] = __builtin_amdgcn_wmma_f32_16x16x32_bf16(
                          false, ah[ks].f, false, bl.f, (short)0, acc[nt], false, false);
            acc[nt] = __builtin_amdgcn_wmma_f32_16x16x32_bf16(
                          false, al[ks].f, false, bh.f, (short)0, acc[nt], false, false);
        }
    }

    // per-lane epilogue constants: 8 consecutive output rows -> vector LDS loads
    const int i0 = wv * 16 + khalf * 8;            // first kernel row of this lane
    unsigned int rowoff[8];
    float bb[8];
    {
        uint4 r0 = *(const uint4*)&sRow[i0];
        uint4 r1 = *(const uint4*)&sRow[i0 + 4];
        rowoff[0]=r0.x; rowoff[1]=r0.y; rowoff[2]=r0.z; rowoff[3]=r0.w;
        rowoff[4]=r1.x; rowoff[5]=r1.y; rowoff[6]=r1.z; rowoff[7]=r1.w;
        float4 b0 = *(const float4*)&sbias[i0];
        float4 b1 = *(const float4*)&sbias[i0 + 4];
        bb[0]=b0.x; bb[1]=b0.y; bb[2]=b0.z; bb[3]=b0.w;
        bb[4]=b1.x; bb[5]=b1.y; bb[6]=b1.z; bb[7]=b1.w;
    }
    const int jn = j0 + nrow;                      // j = jn + nt*16

    // only the LAST position tile can hit the ragged row ends:
    // min row length = LSEQ-127 = 16257 > max j (16255) of every earlier tile
    if (j0 + NTILE + KK - 2 < LSEQ) {              // scalar branch: guard-free path
        #pragma unroll
        for (int nt = 0; nt < 8; ++nt) {
            #pragma unroll
            for (int r = 0; r < 8; ++r) {
                float v = acc[nt][r] + bb[r];
                v = v > 0.0f ? v : 0.0f;
                __builtin_nontemporal_store(
                    v, out + (size_t)rowoff[r] + (unsigned int)(jn + nt * 16));
            }
        }
    } else {                                       // ragged tail tile
        #pragma unroll
        for (int nt = 0; nt < 8; ++nt) {
            #pragma unroll
            for (int r = 0; r < 8; ++r) {
                const int j = jn + nt * 16;
                if (j < LSEQ - (i0 + r)) {
                    float v = acc[nt][r] + bb[r];
                    v = v > 0.0f ? v : 0.0f;
                    __builtin_nontemporal_store(
                        v, out + (size_t)rowoff[r] + (unsigned int)j);
                }
            }
        }
    }
}

extern "C" void kernel_launch(void* const* d_in, const int* in_sizes, int n_in,
                              void* d_out, int out_size, void* d_ws, size_t ws_size,
                              hipStream_t stream) {
    const float* x = (const float*)d_in[0];
    bf16_t* Wh = (bf16_t*)d_ws;
    bf16_t* Wl = Wh + KK * KK;        // 64 KB total scratch

    const float* biases;
    if (n_in >= KK + 2) {
        // weights as 128 separate tensors: d_in[1..128], biases d_in[129]
        WPtrs wp;
        for (int i = 0; i < KK; ++i) wp.w[i] = (const float*)d_in[1 + i];
        biases = (const float*)d_in[KK + 1];
        build_w_ptrs<<<dim3(KK), dim3(KK), 0, stream>>>(wp, Wh, Wl);
    } else {
        // weights concatenated into one buffer at d_in[1]
        biases = (const float*)d_in[2];
        build_w_concat<<<dim3(KK), dim3(KK), 0, stream>>>((const float*)d_in[1], Wh, Wl);
    }

    dim3 grid(LSEQ / NTILE, BATCH);   // 128 x 64 workgroups, 256 threads (8 waves)
    conv_wmma<<<grid, 256, 0, stream>>>(x, biases, Wh, Wl, (float*)d_out);
}
